// Model_48670569398564
// MI455X (gfx1250) — matrix operand order repaired and checked by reading, not compiled, not run on hardware
//
#include <hip/hip_runtime.h>

#define NN 50000
#define EE 1600000
#define H 128
#define R 8
#define TT 2
#define PP 1000
#define MM 223

typedef __attribute__((ext_vector_type(16))) _Float16 v16h;
typedef __attribute__((ext_vector_type(8)))  float    v8f;

union Frag16 { v16h v; uint4 q[2]; };
union H4     { uint2 u; _Float16 h[4]; };

__device__ __forceinline__ float wave_reduce_sum(float v) {
    #pragma unroll
    for (int m = 16; m >= 1; m >>= 1) v += __shfl_xor(v, m, 32);
    return v;
}

// ---------------- zero fill ----------------
__global__ void kz_zero(float* p, size_t n) {
    size_t i = (size_t)blockIdx.x * blockDim.x + threadIdx.x;
    size_t stride = (size_t)gridDim.x * blockDim.x;
    for (; i < n; i += stride) p[i] = 0.f;
}

// ---------------- K0: convert + pre-swizzle weights into WMMA B-fragment layout ----
// w_rel_sw: [r][kb][ct][lane][i]  (R*4*8*32*16 halves)
//   element i of lane's B frag = B[kb*32 + (lane>>4)*16 + i][ct*16 + (lane&15)]
// fl3_sw:  [kb][ct][lane][i]     (4*64*32*16 halves, cols >= 1000 zero-padded)
__global__ void k0_swz(const float* __restrict__ w_rel, const float* __restrict__ fl3_w,
                       _Float16* __restrict__ w_rel_sw, _Float16* __restrict__ fl3_sw) {
    int idx = blockIdx.x * blockDim.x + threadIdx.x;
    if (idx < R * 4 * 8 * 512) {
        int i    = idx & 15;
        int lane = (idx >> 4) & 31;
        int ct   = (idx >> 9) & 7;
        int kb   = (idx >> 12) & 3;
        int r    = idx >> 15;
        int hlf = lane >> 4, l = lane & 15;
        int k   = kb * 32 + hlf * 16 + i;
        int col = ct * 16 + l;
        w_rel_sw[idx] = (_Float16)w_rel[((size_t)r * H + k) * H + col];
    }
    if (idx < 4 * 64 * 512) {
        int i    = idx & 15;
        int lane = (idx >> 4) & 31;
        int ct   = (idx >> 9) & 63;
        int kb   = idx >> 15;
        int hlf = lane >> 4, l = lane & 15;
        int k   = kb * 32 + hlf * 16 + i;
        int col = ct * 16 + l;
        fl3_sw[idx] = (col < PP) ? (_Float16)fl3_w[(size_t)k * PP + col] : (_Float16)0.f;
    }
}

// ---------------- K1: h0 = ntc@fc_w+b ; h_nt = h0 @ w_nt[node_type] ----------------
__global__ void k1_hnt(const float* __restrict__ ntc, const float* __restrict__ fc_w,
                       const float* __restrict__ fc_b, const float* __restrict__ w_nt,
                       const int* __restrict__ node_type, _Float16* __restrict__ h_nt) {
    __shared__ float h0[H];
    int n = blockIdx.x;
    int j = threadIdx.x;                       // 0..127
    float a0 = ntc[2 * n], a1 = ntc[2 * n + 1];
    h0[j] = a0 * fc_w[j] + a1 * fc_w[H + j] + fc_b[j];
    __syncthreads();
    int t = node_type[n];
    const float* W = w_nt + (size_t)t * H * H;
    float acc = 0.f;
    #pragma unroll 8
    for (int i = 0; i < H; ++i) acc += h0[i] * W[i * H + j];
    h_nt[(size_t)n * H + j] = (_Float16)acc;
}

// ---------------- K2: h_rel[n,r,:] = h_nt[n,:] @ w_rel[r]  (f16 WMMA, f32 accum) ----
// One block per 16-node tile; wave = output col tile; A frags loaded once, reused
// across all 8 relations; B frags are 2x b128 from pre-swizzled layout.
__global__ void k2_hrel(const _Float16* __restrict__ h_nt,
                        const _Float16* __restrict__ w_rel_sw,
                        _Float16* __restrict__ h_rel) {
    int tile = blockIdx.x;             // 16 nodes
    int ct   = threadIdx.x >> 5;       // col tile 0..7
    int lane = threadIdx.x & 31;
    int hlf  = lane >> 4;
    int l    = lane & 15;
    int row  = tile * 16 + l;
    Frag16 a[4];
    #pragma unroll
    for (int kb = 0; kb < 4; ++kb) {
        const uint4* pa = (const uint4*)(h_nt + (size_t)row * H + kb * 32 + hlf * 16);
        a[kb].q[0] = pa[0];           // K = kb*32 + hlf*16     .. +7
        a[kb].q[1] = pa[1];           // K = kb*32 + hlf*16 + 8 .. +15
    }
    for (int r = 0; r < R; ++r) {
        v8f c = {};
        #pragma unroll
        for (int kb = 0; kb < 4; ++kb) {
            Frag16 b;
            const uint4* pb = (const uint4*)(w_rel_sw +
                ((size_t)(((r * 4 + kb) * 8 + ct) * 32 + lane)) * 16);
            b.q[0] = pb[0];
            b.q[1] = pb[1];
            c = __builtin_amdgcn_wmma_f32_16x16x32_f16(false, a[kb].v, false, b.v,
                                                       (short)0, c, false, false);
        }
        #pragma unroll
        for (int v = 0; v < 8; ++v) {
            int node = tile * 16 + hlf * 8 + v;
            h_rel[((size_t)node * R + r) * H + ct * 16 + l] = (_Float16)c[v];
        }
    }
}

// ---------------- K3: edge gather * norm, scatter-add to h[dst] ----------------
__global__ void k3_edges(const _Float16* __restrict__ h_rel,
                         const float* __restrict__ norm,
                         const int* __restrict__ src, const int* __restrict__ dst,
                         const int* __restrict__ etype, float* __restrict__ h) {
    int e = blockIdx.x * 8 + (threadIdx.x >> 5);
    if (e >= EE) return;
    int lane = threadIdx.x & 31;
    int s = src[e], d = dst[e], et = etype[e];
    float nm = norm[e];
    H4 val;
    val.u = *(const uint2*)(h_rel + ((size_t)s * R + et) * H + lane * 4);
    float* hp = h + (size_t)d * H + lane * 4;
    atomicAdd(hp + 0, (float)val.h[0] * nm);
    atomicAdd(hp + 1, (float)val.h[1] * nm);
    atomicAdd(hp + 2, (float)val.h[2] * nm);
    atomicAdd(hp + 3, (float)val.h[3] * nm);
}

// ---------------- K4a: relu in place, f16 copy, gembed partial sums ----------------
__global__ void k4a_relu(float* h, _Float16* h_h, float* gembed) {
    int j = threadIdx.x;               // 0..127
    int base = blockIdx.x * 64;
    float g = 0.f;
    for (int k = 0; k < 64; ++k) {
        int n = base + k;
        if (n >= NN) break;
        size_t idx = (size_t)n * H + j;
        float v = fmaxf(h[idx], 0.f);
        h[idx] = v;
        h_h[idx] = (_Float16)v;
        g += v;
    }
    atomicAdd(&gembed[j], g);
}

// ---------------- K4b: pred = clamp(sigmoid(h @ fc2_w + b)) ----------------
__global__ void k4b_pred(const float* __restrict__ h, const float* __restrict__ fc2_w,
                         const float* __restrict__ fc2_b, float* __restrict__ pred) {
    int n = blockIdx.x * 8 + (threadIdx.x >> 5);
    if (n >= NN) return;
    int lane = threadIdx.x & 31;
    const float* hp = h + (size_t)n * H + lane * 4;
    const float* wp = fc2_w + lane * 4;
    float s = hp[0] * wp[0] + hp[1] * wp[1] + hp[2] * wp[2] + hp[3] * wp[3];
    s = wave_reduce_sum(s);
    if (lane == 0) {
        float x = s + fc2_b[0];
        float sg = 1.f / (1.f + __expf(-x));
        pred[n] = fminf(fmaxf(sg, 1e-7f), 1e10f);
    }
}

// ---------------- K5: prob3_sum = sum(sigmoid(h @ fl3_w + b))  (f16 WMMA) ----------
// One block per node tile; A frags loaded once per wave, 8 col tiles per wave.
__global__ void k5_prob3(const _Float16* __restrict__ h_h,
                         const _Float16* __restrict__ fl3_sw,
                         const float* __restrict__ fl3_b,
                         float* __restrict__ prob3_acc) {
    __shared__ float part[8];
    int tile = blockIdx.x;
    int wv   = threadIdx.x >> 5;
    int lane = threadIdx.x & 31;
    int hlf  = lane >> 4;
    int l    = lane & 15;
    int row  = tile * 16 + l;
    Frag16 a[4];
    #pragma unroll
    for (int kb = 0; kb < 4; ++kb) {
        const uint4* pa = (const uint4*)(h_h + (size_t)row * H + kb * 32 + hlf * 16);
        a[kb].q[0] = pa[0];
        a[kb].q[1] = pa[1];
    }
    float s = 0.f;
    for (int cti = 0; cti < 8; ++cti) {
        int ct = wv * 8 + cti;             // 0..63 (1024 padded cols)
        v8f c = {};
        #pragma unroll
        for (int kb = 0; kb < 4; ++kb) {
            Frag16 b;
            const uint4* pb = (const uint4*)(fl3_sw +
                ((size_t)((kb * 64 + ct) * 32 + lane)) * 16);
            b.q[0] = pb[0];
            b.q[1] = pb[1];
            c = __builtin_amdgcn_wmma_f32_16x16x32_f16(false, a[kb].v, false, b.v,
                                                       (short)0, c, false, false);
        }
        int col = ct * 16 + l;
        if (col < PP) {
            float bb = fl3_b[col];
            #pragma unroll
            for (int v = 0; v < 8; ++v) {
                float x = c[v] + bb;
                s += 1.f / (1.f + __expf(-x));
            }
        }
    }
    s = wave_reduce_sum(s);
    if (lane == 0) part[wv] = s;
    __syncthreads();
    if (threadIdx.x == 0) {
        float t = 0.f;
        #pragma unroll
        for (int w = 0; w < 8; ++w) t += part[w];
        atomicAdd(prob3_acc, t);
    }
}

// ---------------- K6a: row sums + ce + sum(pm^2) ----------------
__global__ void k6_rows(const float* __restrict__ pred, const float* __restrict__ gt,
                        float* __restrict__ rs, float* __restrict__ sc) {
    __shared__ float rr[256], ee[256], ss[256];
    int row = blockIdx.x;
    int t = threadIdx.x;
    float rsum = 0.f, ce = 0.f, s2 = 0.f;
    if (t < MM) {
        float pm = pred[row * MM + t];
        rsum = pm;
        ce = -gt[row * MM + t] * __log2f(pm);
        s2 = pm * pm;
    }
    rr[t] = rsum; ee[t] = ce; ss[t] = s2;
    __syncthreads();
    for (int s = 128; s > 0; s >>= 1) {
        if (t < s) { rr[t] += rr[t + s]; ee[t] += ee[t + s]; ss[t] += ss[t + s]; }
        __syncthreads();
    }
    if (t == 0) {
        rs[row] = rr[0];
        atomicAdd(&sc[0], ee[0]);   // ce
        atomicAdd(&sc[1], ss[0]);   // sum pm^2
    }
}

// ---------------- K6b: col sums ----------------
__global__ void k6_cols(const float* __restrict__ pred, float* __restrict__ cs) {
    __shared__ float rr[256];
    int col = blockIdx.x;
    int t = threadIdx.x;
    rr[t] = (t < MM) ? pred[t * MM + col] : 0.f;
    __syncthreads();
    for (int s = 128; s > 0; s >>= 1) {
        if (t < s) rr[t] += rr[t + s];
        __syncthreads();
    }
    if (t == 0) cs[col] = rr[0];
}

// ---------------- K7: agg = segment_sum(pred[src], dst) ----------------
__global__ void k7_agg(const float* __restrict__ pred, const int* __restrict__ src,
                       const int* __restrict__ dst, float* __restrict__ agg) {
    int e = blockIdx.x * blockDim.x + threadIdx.x;
    if (e < EE) atomicAdd(&agg[dst[e]], pred[src[e]]);
}

// ---------------- K8: out_cost = sum((agg-1)^2 * special_cost) ----------------
__global__ void k8_outcost(const float* __restrict__ agg, const float* __restrict__ scost,
                           float* __restrict__ sc) {
    __shared__ float rr[256];
    int n = blockIdx.x * 256 + threadIdx.x;
    float v = 0.f;
    if (n < NN) { float d = agg[n] - 1.f; v = d * d * scost[n]; }
    rr[threadIdx.x] = v;
    __syncthreads();
    for (int s = 128; s > 0; s >>= 1) {
        if (threadIdx.x < s) rr[threadIdx.x] += rr[threadIdx.x + s];
        __syncthreads();
    }
    if (threadIdx.x == 0) atomicAdd(&sc[3], rr[0]);
}

// ---------------- K9: assemble the 6 outputs ----------------
__global__ void k9_final(const float* __restrict__ gembed, const float* __restrict__ fc3_w,
                         const float* __restrict__ fc3_b, const float* __restrict__ rs,
                         const float* __restrict__ cs, const float* __restrict__ sc,
                         float* __restrict__ out) {
    __shared__ float A[256], B[256], C[256];
    int t = threadIdx.x;
    float a = (t < H) ? gembed[t] * (1.f / (float)NN) * fc3_w[t] : 0.f;
    float b2 = 0.f, c2 = 0.f;
    if (t < MM) {
        float r = rs[t], c = cs[t];
        b2 = r * r + c * c;
        c2 = (r - 1.f) * (r - 1.f) + (c - 1.f) * (c - 1.f);
    }
    A[t] = a; B[t] = b2; C[t] = c2;
    __syncthreads();
    for (int s = 128; s > 0; s >>= 1) {
        if (t < s) { A[t] += A[t + s]; B[t] += B[t + s]; C[t] += C[t + s]; }
        __syncthreads();
    }
    if (t == 0) {
        out[0] = A[0] + fc3_b[0];          // pred_obj
        out[1] = C[0];                     // sumone
        out[2] = B[0] - 2.f * sc[1];       // loss2 = sum(rs^2)+sum(cs^2)-2*sum(pm^2)
        out[3] = sc[0];                    // ce
        out[4] = sc[3];                    // out_cost
        out[5] = sc[2];                    // prob3_sum
    }
}

extern "C" void kernel_launch(void* const* d_in, const int* in_sizes, int n_in,
                              void* d_out, int out_size, void* d_ws, size_t ws_size,
                              hipStream_t stream) {
    const float* ntc    = (const float*)d_in[0];
    const float* norm   = (const float*)d_in[1];
    const float* scost  = (const float*)d_in[2];
    const float* gt_map = (const float*)d_in[3];
    const float* fc_w   = (const float*)d_in[4];
    const float* fc_b   = (const float*)d_in[5];
    const float* w_nt   = (const float*)d_in[6];
    const float* w_rel  = (const float*)d_in[7];
    const float* fc2_w  = (const float*)d_in[8];
    const float* fc2_b  = (const float*)d_in[9];
    const float* fc3_w  = (const float*)d_in[10];
    const float* fc3_b  = (const float*)d_in[11];
    const float* fl3_w  = (const float*)d_in[12];
    const float* fl3_b  = (const float*)d_in[13];
    const int* node_type = (const int*)d_in[14];
    const int* src       = (const int*)d_in[15];
    const int* dst       = (const int*)d_in[16];
    const int* etype     = (const int*)d_in[17];
    float* out = (float*)d_out;

    char* ws = (char*)d_ws;
    size_t off = 0;
    auto alloc = [&](size_t bytes) -> void* {
        void* p = ws + off;
        off = (off + bytes + 255) & ~(size_t)255;
        return p;
    };
    _Float16* h_nt     = (_Float16*)alloc((size_t)NN * H * 2);        // 12.8 MB
    _Float16* w_rel_sw = (_Float16*)alloc((size_t)R * 4 * 8 * 512 * 2);  // 256 KB
    _Float16* fl3_sw   = (_Float16*)alloc((size_t)4 * 64 * 512 * 2);     // 256 KB
    _Float16* h_rel    = (_Float16*)alloc((size_t)NN * R * H * 2);    // 102.4 MB
    float*    h        = (float*)alloc((size_t)NN * H * 4);           // 25.6 MB
    _Float16* h_h      = (_Float16*)alloc((size_t)NN * H * 2);        // 12.8 MB
    float*    pred     = (float*)alloc((size_t)NN * 4);
    float*    agg      = (float*)alloc((size_t)NN * 4);
    float*    gembed   = (float*)alloc(128 * 4);
    float*    rs       = (float*)alloc(256 * 4);
    float*    cs       = (float*)alloc(256 * 4);
    float*    sc       = (float*)alloc(16 * 4);  // [0]=ce [1]=sum pm^2 [2]=prob3 [3]=out_cost

    // zero accumulators
    kz_zero<<<2048, 256, 0, stream>>>(h, (size_t)NN * H);
    kz_zero<<<128, 256, 0, stream>>>(agg, (size_t)NN);
    kz_zero<<<1, 256, 0, stream>>>(gembed, 128);
    kz_zero<<<1, 64, 0, stream>>>(sc, 16);

    // convert + pre-swizzle weights into WMMA B-fragment layout
    k0_swz<<<(R * 4 * 8 * 512 + 255) / 256, 256, 0, stream>>>(w_rel, fl3_w, w_rel_sw, fl3_sw);

    // node feature pipeline
    k1_hnt<<<NN, H, 0, stream>>>(ntc, fc_w, fc_b, w_nt, node_type, h_nt);

    // main relation GEMM: [N,128] x [8][128,128] -> h_rel (WMMA f16)
    k2_hrel<<<NN / 16, 256, 0, stream>>>(h_nt, w_rel_sw, h_rel);

    // edge gather/scale/scatter-add
    k3_edges<<<EE / 8, 256, 0, stream>>>(h_rel, norm, src, dst, etype, h);

    // relu + f16 copy + gembed partials
    k4a_relu<<<(NN + 63) / 64, H, 0, stream>>>(h, h_h, gembed);

    // pred
    k4b_pred<<<(NN + 7) / 8, 256, 0, stream>>>(h, fc2_w, fc2_b, pred);

    // prob3 GEMM+sigmoid+sum (WMMA f16)
    k5_prob3<<<NN / 16, 256, 0, stream>>>(h_h, fl3_sw, fl3_b, &sc[2]);

    // pm statistics
    k6_rows<<<MM, 256, 0, stream>>>(pred, gt_map, rs, sc);
    k6_cols<<<MM, 256, 0, stream>>>(pred, cs);

    // agg + out_cost
    k7_agg<<<(EE + 255) / 256, 256, 0, stream>>>(pred, src, dst, agg);
    k8_outcost<<<(NN + 255) / 256, 256, 0, stream>>>(agg, scost, sc);

    // final assembly
    k9_final<<<1, 256, 0, stream>>>(gembed, fc3_w, fc3_b, rs, cs, sc, out);
}
